// my_lstm_6347961664096
// MI455X (gfx1250) — matrix operand order repaired and checked by reading, not compiled, not run on hardware
//
#include <hip/hip_runtime.h>

// ---------------------------------------------------------------------------
// LSTM on gfx1250 via v_wmma_f32_16x16x32_bf16.
// Sizes: SEQ=512, B=128, I=512, H=1024, 4 gates -> G=4096 cols per step.
// Per step: gates[128,4096] = [x_s | h] (128x1536, bf16) @ [W_ih | W_hh]^T.
// ---------------------------------------------------------------------------

#define SEQ   512
#define NB    128
#define NI    512
#define NH    1024
#define NG    (4 * NH)

typedef __attribute__((ext_vector_type(16))) __bf16 v16bf;
typedef __attribute__((ext_vector_type(8)))  float  v8f;

union FragBF {
    v16bf    v;
    unsigned u[8];
};

__device__ __forceinline__ unsigned short f2bf_rne(float f) {
    unsigned u = __builtin_bit_cast(unsigned, f);
    u += 0x7FFFu + ((u >> 16) & 1u);           // round to nearest even
    return (unsigned short)(u >> 16);
}
// Pack two fp32 -> packed bf16x2 (RNE), merged with a single v_perm_b32.
__device__ __forceinline__ unsigned pack2(float a, float b) {
    unsigned ua = __builtin_bit_cast(unsigned, a);
    unsigned ub = __builtin_bit_cast(unsigned, b);
    ua += 0x7FFFu + ((ua >> 16) & 1u);
    ub += 0x7FFFu + ((ub >> 16) & 1u);
    // result bytes [0,1] = ua bytes [2,3]; result bytes [2,3] = ub bytes [2,3]
    return __builtin_amdgcn_perm(ub, ua, 0x07060302u);
}
__device__ __forceinline__ float sigm(float x) {
    return 1.0f / (1.0f + __expf(-x));
}
__device__ __forceinline__ float fast_tanh(float x) {
    float e = __expf(2.0f * x);
    return (e - 1.0f) / (e + 1.0f);
}

// ISA 7.12.2: 16-bit A-matrix 16x32 K placement per (vgpr v, lane-half)
__device__ __forceinline__ int k_off_a(int v, int half) {
    return ((v & 4) << 2) + (half << 3) + ((v & 3) << 1);
}
// 16-bit B-matrix 32x16 K placement (same K order as A)
__device__ __forceinline__ int k_off_b(int v, int half) {
    return ((v & 4) << 2) + (half << 3) + ((v & 3) << 1);
}

// ---------------------------------------------------------------------------
// Vectorized fp32 -> bf16 conversion: 8 elements per thread (b128 in/out).
// ---------------------------------------------------------------------------
__global__ void cvt_pack8(const float* __restrict__ in,
                          unsigned short* __restrict__ out, int n8) {
    int i = blockIdx.x * blockDim.x + threadIdx.x;
    if (i >= n8) return;
    const float4* p = (const float4*)in + (size_t)i * 2;
    float4 a = p[0], b = p[1];
    uint4 r;
    r.x = pack2(a.x, a.y);
    r.y = pack2(a.z, a.w);
    r.z = pack2(b.x, b.y);
    r.w = pack2(b.z, b.w);
    ((uint4*)out)[i] = r;
}

__global__ void init_state(const float* __restrict__ h0,
                           const float* __restrict__ c0,
                           float* __restrict__ h_f,
                           float* __restrict__ c,
                           unsigned short* __restrict__ hbf0) {
    int i = blockIdx.x * blockDim.x + threadIdx.x;
    if (i >= NB * NH) return;
    float h = h0[i];
    h_f[i]  = h;
    c[i]    = c0[i];
    hbf0[i] = f2bf_rne(h);
}

// ---------------------------------------------------------------------------
// One recurrent step: GEMM over K=1536 (512 from x_s, 1024 from h bf16),
// 4 gate columns per wave, fused LSTM pointwise update.
// XBF=true : x_s is pre-converted bf16 (pure b128 loads, preferred path)
// XBF=false: x_s is fp32, converted in-register (small-workspace fallback)
// Grid: 64 blocks x 256 threads (8 waves). Wave (mt, jt) owns batch rows
// [16*mt,16*mt+16) and hidden cols [16*jt,16*jt+16) for all four gates.
// ---------------------------------------------------------------------------
template <bool XBF>
__global__ void __launch_bounds__(256) lstm_step(
        const void*           __restrict__ x_s_,   // [128,512] bf16 or fp32
        const unsigned short* __restrict__ h_in,   // [128,1024] bf16
        unsigned short*       __restrict__ h_out,  // [128,1024] bf16
        float*                __restrict__ c,      // [128,1024] fp32 (in/out)
        float*                __restrict__ h_f,    // [128,1024] fp32 (out)
        const unsigned short* __restrict__ WihB,   // [4096,512] bf16 row-major
        const unsigned short* __restrict__ WhhB,   // [4096,1024] bf16 row-major
        const float*          __restrict__ b_ih,   // [4096]
        const float*          __restrict__ b_hh)   // [4096]
{
    const int wave = threadIdx.x >> 5;
    const int lane = threadIdx.x & 31;
    const int half = lane >> 4;
    const int lmod = lane & 15;

    const int mt = blockIdx.x & 7;                    // batch tile 0..7
    const int jt = ((blockIdx.x >> 3) << 3) + wave;   // hidden-col tile 0..63

    const int m_a = mt * 16 + lmod;                   // A-fragment row (batch)
    const int n_b = jt * 16 + lmod;                   // B-fragment col (hidden)

    v8f acc[4];
    #pragma unroll
    for (int g = 0; g < 4; ++g) acc[g] = v8f{};

    // ---- Phase 1: K = 0..511 from x_s --------------------------------------
    for (int kb = 0; kb < NI; kb += 32) {
        FragBF a;
        if (XBF) {
            const unsigned short* xrow =
                (const unsigned short*)x_s_ + (size_t)m_a * NI;
            #pragma unroll
            for (int v = 0; v < 8; ++v) {
                int k = kb + k_off_a(v, half);
                a.u[v] = *(const unsigned*)(xrow + k);
            }
        } else {
            const float* xrow = (const float*)x_s_ + (size_t)m_a * NI;
            #pragma unroll
            for (int v = 0; v < 8; ++v) {
                int k = kb + k_off_a(v, half);
                float2 t = *(const float2*)(xrow + k);
                a.u[v] = pack2(t.x, t.y);
            }
        }
        #pragma unroll
        for (int g = 0; g < 4; ++g) {
            const unsigned short* brow = WihB + (size_t)(g * NH + n_b) * NI;
            FragBF b;
            #pragma unroll
            for (int v = 0; v < 8; ++v) {
                int k = kb + k_off_b(v, half);
                b.u[v] = *(const unsigned*)(brow + k);
            }
            acc[g] = __builtin_amdgcn_wmma_f32_16x16x32_bf16(
                false, a.v, false, b.v, (short)0, acc[g], false, false);
        }
    }

    // ---- Phase 2: K = 0..1023 from h (bf16) --------------------------------
    const unsigned short* hrow = h_in + (size_t)m_a * NH;
    for (int kb = 0; kb < NH; kb += 32) {
        FragBF a;
        #pragma unroll
        for (int v = 0; v < 8; ++v) {
            int k = kb + k_off_a(v, half);
            a.u[v] = *(const unsigned*)(hrow + k);
        }
        #pragma unroll
        for (int g = 0; g < 4; ++g) {
            const unsigned short* brow = WhhB + (size_t)(g * NH + n_b) * NH;
            FragBF b;
            #pragma unroll
            for (int v = 0; v < 8; ++v) {
                int k = kb + k_off_b(v, half);
                b.u[v] = *(const unsigned*)(brow + k);
            }
            acc[g] = __builtin_amdgcn_wmma_f32_16x16x32_bf16(
                false, a.v, false, b.v, (short)0, acc[g], false, false);
        }
    }

    // ---- Fused LSTM pointwise update ---------------------------------------
    // C/D layout: element r of this lane is (m = 16*mt + r + 8*half, n = n_b).
    const float bias_i = b_ih[0 * NH + n_b] + b_hh[0 * NH + n_b];
    const float bias_f = b_ih[1 * NH + n_b] + b_hh[1 * NH + n_b];
    const float bias_g = b_ih[2 * NH + n_b] + b_hh[2 * NH + n_b];
    const float bias_o = b_ih[3 * NH + n_b] + b_hh[3 * NH + n_b];

    #pragma unroll
    for (int r = 0; r < 8; ++r) {
        int m   = mt * 16 + r + (half << 3);
        int idx = m * NH + n_b;
        float iv = sigm(acc[0][r] + bias_i);
        float fv = sigm(acc[1][r] + bias_f);
        float gv = fast_tanh(acc[2][r] + bias_g);
        float ov = sigm(acc[3][r] + bias_o);
        float cn = fv * c[idx] + iv * gv;
        float hn = ov * fast_tanh(cn);
        c[idx]     = cn;
        h_f[idx]   = hn;
        h_out[idx] = f2bf_rne(hn);
    }
}

// ---------------------------------------------------------------------------
// Decoder + softmax over the batch axis (axis 0). One block, 128 threads.
// ---------------------------------------------------------------------------
__global__ void decode_softmax(const float* __restrict__ h_f,
                               const float* __restrict__ W_dec,  // [4,1024]
                               const float* __restrict__ b_dec,  // [4]
                               float* __restrict__ out)          // [128,4]
{
    __shared__ float sm[4][NB];
    __shared__ float red[8];   // [0..3]=max, [4..7]=sum
    int b = threadIdx.x;       // 0..127

    float acc0 = b_dec[0], acc1 = b_dec[1], acc2 = b_dec[2], acc3 = b_dec[3];
    const float* hr = h_f + (size_t)b * NH;
    for (int k = 0; k < NH; ++k) {
        float h = hr[k];
        acc0 += h * W_dec[0 * NH + k];
        acc1 += h * W_dec[1 * NH + k];
        acc2 += h * W_dec[2 * NH + k];
        acc3 += h * W_dec[3 * NH + k];
    }
    sm[0][b] = acc0; sm[1][b] = acc1; sm[2][b] = acc2; sm[3][b] = acc3;
    __syncthreads();

    if (b < 4) {
        float mx = -3.4e38f;
        for (int i = 0; i < NB; ++i) mx = fmaxf(mx, sm[b][i]);
        float s = 0.0f;
        for (int i = 0; i < NB; ++i) s += __expf(sm[b][i] - mx);
        red[b]     = mx;
        red[4 + b] = s;
    }
    __syncthreads();

    #pragma unroll
    for (int o = 0; o < 4; ++o)
        out[b * 4 + o] = __expf(sm[o][b] - red[o]) / red[4 + o];
}

// ---------------------------------------------------------------------------
// Host side
// ---------------------------------------------------------------------------
extern "C" void kernel_launch(void* const* d_in, const int* in_sizes, int n_in,
                              void* d_out, int out_size, void* d_ws, size_t ws_size,
                              hipStream_t stream) {
    (void)in_sizes; (void)n_in; (void)out_size;

    const float* inputs = (const float*)d_in[0];   // [512,128,512]
    const float* h0     = (const float*)d_in[1];   // [128,1024]
    const float* c0     = (const float*)d_in[2];   // [128,1024]
    const float* W_ih   = (const float*)d_in[3];   // [4096,512]
    const float* W_hh   = (const float*)d_in[4];   // [4096,1024]
    const float* b_ih   = (const float*)d_in[5];   // [4096]
    const float* b_hh   = (const float*)d_in[6];   // [4096]
    const float* W_dec  = (const float*)d_in[7];   // [4,1024]
    const float* b_dec  = (const float*)d_in[8];   // [4]
    float*       out    = (float*)d_out;           // [128,4]

    // Workspace layout
    char* w = (char*)d_ws;
    unsigned short* WihB = (unsigned short*)(w + 0);            //  4 MiB
    unsigned short* WhhB = (unsigned short*)(w + 4194304);      //  8 MiB
    unsigned short* hbf  = (unsigned short*)(w + 12582912);     //  2x256 KiB
    float*          h_f  = (float*)(w + 13107200);              //  512 KiB
    float*          c    = (float*)(w + 13631488);              //  512 KiB
    unsigned short* xbf  = (unsigned short*)(w + 14155776);     // 64 MiB (optional)
    const size_t NEED_BIG = 14155776ull + (size_t)SEQ * NB * NI * 2; // ~77.5 MiB
    const bool big = (ws_size >= NEED_BIG);

    // 1) weights -> bf16 (8 elements per thread)
    cvt_pack8<<<(NG * NI / 8 + 255) / 256, 256, 0, stream>>>(W_ih, WihB, NG * NI / 8);
    cvt_pack8<<<(NG * NH / 8 + 255) / 256, 256, 0, stream>>>(W_hh, WhhB, NG * NH / 8);

    // 2) inputs -> bf16 upfront (bandwidth-bound, fully parallel) if ws allows
    if (big) {
        int n8 = SEQ * NB * NI / 8;   // 4,194,304
        cvt_pack8<<<(n8 + 255) / 256, 256, 0, stream>>>(inputs, xbf, n8);
    }

    // 3) state init (re-done every call -> deterministic)
    init_state<<<(NB * NH) / 256, 256, 0, stream>>>(h0, c0, h_f, c, hbf);

    // 4) 512 sequential recurrent steps (h bf16 ping-pong to avoid WAR race)
    for (int s = 0; s < SEQ; ++s) {
        unsigned short* h_in = hbf + (size_t)(s & 1) * NB * NH;
        unsigned short* h_o  = hbf + (size_t)((s + 1) & 1) * NB * NH;
        if (big) {
            const unsigned short* x_s = xbf + (size_t)s * NB * NI;
            lstm_step<true><<<64, 256, 0, stream>>>(x_s, h_in, h_o, c, h_f,
                                                    WihB, WhhB, b_ih, b_hh);
        } else {
            const float* x_s = inputs + (size_t)s * NB * NI;
            lstm_step<false><<<64, 256, 0, stream>>>(x_s, h_in, h_o, c, h_f,
                                                     WihB, WhhB, b_ih, b_hh);
        }
    }

    // 5) decoder + softmax(dim=0)
    decode_softmax<<<1, 128, 0, stream>>>(h_f, W_dec, b_dec, out);
}
